// RoPECrossAttention_54443005444919
// MI455X (gfx1250) — matrix operand order repaired and checked by reading, not compiled, not run on hardware
//
#include <hip/hip_runtime.h>

typedef _Float16 half_t;
typedef __attribute__((ext_vector_type(16))) _Float16 v16h;
typedef __attribute__((ext_vector_type(8)))  _Float16 v8h;
typedef __attribute__((ext_vector_type(8)))  float    v8f;
typedef __attribute__((ext_vector_type(4)))  unsigned int u32x4;
typedef __attribute__((ext_vector_type(8)))  int          i32x8;
typedef __attribute__((ext_vector_type(4)))  int          i32x4;

#define D_MODEL  1024
#define NHEAD    16
#define HEAD_DIM 64

__device__ __forceinline__ v8f wmma_f16(v16h a, v16h b, v8f c) {
  // D = A(16x32 f16) * B(32x16 f16) + C(16x16 f32)
  return __builtin_amdgcn_wmma_f32_16x16x32_f16(
      /*neg_a=*/false, a, /*neg_b=*/false, b,
      /*c_mod=*/(short)0, c, /*reuse_a=*/false, /*reuse_b=*/false);
}

__device__ __forceinline__ float half16_max(float v) {
  #pragma unroll
  for (int off = 1; off < 16; off <<= 1)
    v = fmaxf(v, __shfl_xor(v, off, 32));
  return v;
}
__device__ __forceinline__ float half16_sum(float v) {
  #pragma unroll
  for (int off = 1; off < 16; off <<= 1)
    v += __shfl_xor(v, off, 32);
  return v;
}

// ---------------------------------------------------------------------------
// GEMM: Y = X @ W^T + b, X [Mrows,1024] f32, W [1024,1024] f32.
// Output f16 in [B, H, L, 64] layout; optional RoPE (rope_stride != 0).
// Tile: 128x64x32, 256 threads = 8 waves, each wave 2x2 WMMA.
// Software pipelined: next tile staged in registers while WMMAs run.
// ---------------------------------------------------------------------------
__global__ __launch_bounds__(256)
void proj_rope_kernel(const float* __restrict__ X, const float* __restrict__ W,
                      const float* __restrict__ bias, half_t* __restrict__ Y,
                      int L, int rope_stride)
{
  __shared__ __align__(128) half_t As[128][32];
  __shared__ __align__(128) half_t Bs[64][32];
  __shared__ __align__(128) float  Cs[128][64];

  const int t    = threadIdx.x;
  const int wave = t >> 5;
  const int lane = t & 31;
  const int l16  = lane & 15;
  const int hh   = lane >> 4;
  const int m0   = blockIdx.x * 128;
  const int n0   = blockIdx.y * 64;
  const int wm   = (wave >> 1) * 32;
  const int wn   = (wave & 1) * 32;

  v8f zero = {};
  v8f acc[2][2];
  #pragma unroll
  for (int i = 0; i < 2; i++)
    #pragma unroll
    for (int j = 0; j < 2; j++) acc[i][j] = zero;

  float4 xa[4], xb[2];
  auto load_tiles = [&](int k0) {
    #pragma unroll
    for (int i = 0; i < 4; i++) {
      int c = t + i * 256, r = c >> 3, cc = (c & 7) << 2;
      xa[i] = *(const float4*)(X + (size_t)(m0 + r) * D_MODEL + k0 + cc);
    }
    #pragma unroll
    for (int i = 0; i < 2; i++) {
      int c = t + i * 256, r = c >> 3, cc = (c & 7) << 2;
      xb[i] = *(const float4*)(W + (size_t)(n0 + r) * D_MODEL + k0 + cc);
    }
  };
  load_tiles(0);

  for (int k0 = 0; k0 < D_MODEL; k0 += 32) {
    __syncthreads();  // previous tile's WMMAs done -> LDS reusable
    #pragma unroll
    for (int i = 0; i < 4; i++) {
      int c = t + i * 256, r = c >> 3, cc = (c & 7) << 2;
      As[r][cc + 0] = (half_t)xa[i].x; As[r][cc + 1] = (half_t)xa[i].y;
      As[r][cc + 2] = (half_t)xa[i].z; As[r][cc + 3] = (half_t)xa[i].w;
    }
    #pragma unroll
    for (int i = 0; i < 2; i++) {
      int c = t + i * 256, r = c >> 3, cc = (c & 7) << 2;
      Bs[r][cc + 0] = (half_t)xb[i].x; Bs[r][cc + 1] = (half_t)xb[i].y;
      Bs[r][cc + 2] = (half_t)xb[i].z; Bs[r][cc + 3] = (half_t)xb[i].w;
    }
    if (k0 + 32 < D_MODEL) load_tiles(k0 + 32);  // overlap with compute below
    __syncthreads();

    #pragma unroll
    for (int ti = 0; ti < 2; ti++) {
      const half_t* ar = &As[wm + ti * 16 + l16][0];
      v8h alo = *(const v8h*)(ar + hh * 16);
      v8h ahi = *(const v8h*)(ar + hh * 16 + 8);
      v16h a;
      #pragma unroll
      for (int j = 0; j < 8; j++) { a[j] = alo[j]; a[8 + j] = ahi[j]; }
      #pragma unroll
      for (int tj = 0; tj < 2; tj++) {
        v16h b = *(const v16h*)(&Bs[wn + tj * 16 + l16][hh * 16]);
        acc[ti][tj] = wmma_f16(a, b, acc[ti][tj]);
      }
    }
  }
  __syncthreads();

  // stage accumulators to LDS (C layout: M = r + 8*hh, N = l16)
  #pragma unroll
  for (int ti = 0; ti < 2; ti++)
    #pragma unroll
    for (int tj = 0; tj < 2; tj++)
      #pragma unroll
      for (int r = 0; r < 8; r++)
        Cs[wm + ti * 16 + r + 8 * hh][wn + tj * 16 + l16] = acc[ti][tj][r];
  __syncthreads();

  // epilogue: bias + optional RoPE, scatter to [B, H, L, 64] f16
  const int head = blockIdx.y;
  for (int idx = t; idx < 128 * 64; idx += 256) {
    int mr   = idx >> 6;
    int d    = idx & 63;
    int grow = m0 + mr;
    int bi   = grow / L;
    int l    = grow - bi * L;
    float y  = Cs[mr][d] + bias[n0 + d];
    float ov = y;
    if (rope_stride != 0) {
      float part = Cs[mr][d ^ 32] + bias[n0 + (d ^ 32)];
      float rot  = (d < 32) ? -part : part;
      float ang  = (float)(l * rope_stride) *
                   __powf(10000.0f, -(float)(d & 31) * (1.0f / 32.0f));
      float sn, cs;
      __sincosf(ang, &sn, &cs);
      ov = y * cs + rot * sn;
    }
    Y[(((size_t)bi * NHEAD + head) * L + l) * HEAD_DIM + d] = (half_t)ov;
  }
}

// ---------------------------------------------------------------------------
// Flash attention. K tile (8 KB contiguous f16) is DMA'd into LDS by the
// Tensor Data Mover (wave 0 issues, waits on TENSORcnt); V tile staged in
// registers and stored transposed; next V tile loads overlap WMMA compute.
// ---------------------------------------------------------------------------
__global__ __launch_bounds__(128)
void attn_kernel(const half_t* __restrict__ Q, const half_t* __restrict__ Kh,
                 const half_t* __restrict__ Vh, half_t* __restrict__ Out,
                 int S, int Klen)
{
  __shared__ __align__(128) half_t Ks[64][64];   // [key][d], filled by TDM
  __shared__ __align__(128) half_t Vt[64][64];   // [d][key] (transposed)
  __shared__ __align__(128) half_t Ps[64][64];   // probabilities [row][key]

  const int t    = threadIdx.x;
  const int wave = t >> 5;
  const int lane = t & 31;
  const int l16  = lane & 15;
  const int hh   = lane >> 4;
  const int sblk = blockIdx.x;
  const int head = blockIdx.y;
  const int b    = blockIdx.z;
  const int wrow = wave * 16;

  const size_t qbase = (((size_t)b * NHEAD + head) * S + sblk * 64) * HEAD_DIM;
  const size_t kbase = (((size_t)b * NHEAD + head) * Klen) * HEAD_DIM;
  const unsigned lds_ks = (unsigned)(uintptr_t)&Ks[0][0];

  // Q fragments, pre-scaled by 1/sqrt(64); held in registers the whole loop
  v16h qa[2];
  #pragma unroll
  for (int f = 0; f < 2; f++) {
    const half_t* qr = Q + qbase + (size_t)(wrow + l16) * HEAD_DIM + f * 32;
    v8h lo = *(const v8h*)(qr + hh * 16);
    v8h hi = *(const v8h*)(qr + hh * 16 + 8);
    #pragma unroll
    for (int j = 0; j < 8; j++) {
      qa[f][j]     = (half_t)((float)lo[j] * 0.125f);
      qa[f][8 + j] = (half_t)((float)hi[j] * 0.125f);
    }
  }

  v8f zero = {};
  v8f O[4];
  #pragma unroll
  for (int di = 0; di < 4; di++) O[di] = zero;
  float mrow[8], lrow[8];
  #pragma unroll
  for (int r = 0; r < 8; r++) { mrow[r] = -1e30f; lrow[r] = 0.0f; }

  v8h vreg[4];
  auto load_v = [&](int kc) {
    #pragma unroll
    for (int i = 0; i < 4; i++) {
      int c = t + i * 128, r = c >> 3, cc = (c & 7) << 3;
      vreg[i] = *(const v8h*)(Vh + kbase + (size_t)(kc + r) * HEAD_DIM + cc);
    }
  };
  load_v(0);

  for (int kc = 0; kc < Klen; kc += 64) {
    __syncthreads();  // previous iteration done reading Ks/Vt/Ps

    if (wave == 0) {
      // TDM: 1-D copy of the 8 KB K tile straight into LDS.
      // data_size=8B; tile_dim0 = tensor_dim0 = 1024 units; 1 row.
      const unsigned long long ga =
          (unsigned long long)(uintptr_t)((const char*)(Kh + kbase) +
                                          (size_t)kc * (HEAD_DIM * 2));
      u32x4 g0;
      g0.x = 1u;                                            // count=1 (valid)
      g0.y = lds_ks;                                        // LDS byte addr
      g0.z = (unsigned)(ga & 0xFFFFFFFFu);                  // global addr lo
      g0.w = (unsigned)((ga >> 32) & 0x01FFFFFFu) | (2u << 30);  // hi | type=2
      i32x8 g1;
      g1[0] = 3 << 16;      // data_size = 3 (8-byte units)
      g1[1] = 1024 << 16;   // tensor_dim0 = 1024 units
      g1[2] = 1 << 16;      // tensor_dim1 = 1
      g1[3] = 1024 << 16;   // tile_dim0 = 1024 units
      g1[4] = 1;            // tile_dim1 = 1
      g1[5] = 1024;         // tensor_dim0_stride = 1024 units
      g1[6] = 0;
      g1[7] = 0;
      i32x4 gz4 = {0, 0, 0, 0};
      i32x8 gz8 = {0, 0, 0, 0, 0, 0, 0, 0};
      __builtin_amdgcn_tensor_load_to_lds(g0, g1, gz4, gz4, gz8, 0);
    }

    // V tile: store registers transposed into LDS
    #pragma unroll
    for (int i = 0; i < 4; i++) {
      int c = t + i * 128, r = c >> 3, cc = (c & 7) << 3;
      #pragma unroll
      for (int j = 0; j < 8; j++) Vt[cc + j][r] = vreg[i][j];
    }
    if (kc + 64 < Klen) load_v(kc + 64);  // overlap with compute below
    if (wave == 0) __builtin_amdgcn_s_wait_tensorcnt(0);
    __syncthreads();

    // scores: 4 tiles of 16 keys, contraction d=64 -> 2 chained WMMAs each
    v8f sc[4];
    #pragma unroll
    for (int ni = 0; ni < 4; ni++) {
      const half_t* kr = &Ks[ni * 16 + l16][0];  // lane = key column of B
      v16h b0 = *(const v16h*)(kr + hh * 16);
      v16h b1 = *(const v16h*)(kr + 32 + hh * 16);
      v8f c = zero;
      c = wmma_f16(qa[0], b0, c);
      c = wmma_f16(qa[1], b1, c);
      sc[ni] = c;
    }

    // online softmax per row (row lives in a 16-lane half-wave)
    #pragma unroll
    for (int r = 0; r < 8; r++) {
      float tmax = -1e30f;
      #pragma unroll
      for (int ni = 0; ni < 4; ni++) tmax = fmaxf(tmax, sc[ni][r]);
      tmax = half16_max(tmax);
      float mnew  = fmaxf(mrow[r], tmax);
      float alpha = __expf(mrow[r] - mnew);
      float rsum  = 0.0f;
      #pragma unroll
      for (int ni = 0; ni < 4; ni++) {
        float p = __expf(sc[ni][r] - mnew);
        sc[ni][r] = p;
        rsum += p;
      }
      rsum = half16_sum(rsum);
      lrow[r] = lrow[r] * alpha + rsum;
      mrow[r] = mnew;
      #pragma unroll
      for (int di = 0; di < 4; di++) O[di][r] *= alpha;
    }

    // write P (f16) for the PV A-fragments
    #pragma unroll
    for (int ni = 0; ni < 4; ni++)
      #pragma unroll
      for (int r = 0; r < 8; r++)
        Ps[wrow + r + 8 * hh][ni * 16 + l16] = (half_t)sc[ni][r];
    __syncthreads();

    // PV: A = P (16x64 strip), B = V (from transposed LDS)
    v16h pa[2];
    #pragma unroll
    for (int f = 0; f < 2; f++) {
      const half_t* pr = &Ps[wrow + l16][f * 32];
      v8h lo = *(const v8h*)(pr + hh * 16);
      v8h hi = *(const v8h*)(pr + hh * 16 + 8);
      #pragma unroll
      for (int j = 0; j < 8; j++) { pa[f][j] = lo[j]; pa[f][8 + j] = hi[j]; }
    }
    #pragma unroll
    for (int di = 0; di < 4; di++) {
      const half_t* vr = &Vt[di * 16 + l16][0];  // lane = d column of B
      v16h b0 = *(const v16h*)(vr + hh * 16);
      v16h b1 = *(const v16h*)(vr + 32 + hh * 16);
      O[di] = wmma_f16(pa[0], b0, O[di]);
      O[di] = wmma_f16(pa[1], b1, O[di]);
    }
  }

  // finalize and store f16 into [B, S, D_MODEL] (col = head*64 + d)
  #pragma unroll
  for (int r = 0; r < 8; r++) {
    float inv = 1.0f / lrow[r];
    #pragma unroll
    for (int di = 0; di < 4; di++) O[di][r] *= inv;
  }
  #pragma unroll
  for (int di = 0; di < 4; di++)
    #pragma unroll
    for (int r = 0; r < 8; r++) {
      int srow = sblk * 64 + wrow + r + 8 * hh;
      int d    = di * 16 + l16;
      Out[((size_t)b * S + srow) * D_MODEL + head * HEAD_DIM + d] =
          (half_t)O[di][r];
    }
}

// ---------------------------------------------------------------------------
// Output projection: Y = Xf16 @ W^T + b, f32 output straight to d_out.
// ---------------------------------------------------------------------------
__global__ __launch_bounds__(256)
void out_proj_kernel(const half_t* __restrict__ X, const float* __restrict__ W,
                     const float* __restrict__ bias, float* __restrict__ Y)
{
  __shared__ __align__(128) half_t As[128][32];
  __shared__ __align__(128) half_t Bs[64][32];

  const int t    = threadIdx.x;
  const int wave = t >> 5;
  const int lane = t & 31;
  const int l16  = lane & 15;
  const int hh   = lane >> 4;
  const int m0   = blockIdx.x * 128;
  const int n0   = blockIdx.y * 64;
  const int wm   = (wave >> 1) * 32;
  const int wn   = (wave & 1) * 32;

  v8f zero = {};
  v8f acc[2][2];
  #pragma unroll
  for (int i = 0; i < 2; i++)
    #pragma unroll
    for (int j = 0; j < 2; j++) acc[i][j] = zero;

  v8h areg[2];
  float4 breg[2];
  auto load_tiles = [&](int k0) {
    #pragma unroll
    for (int i = 0; i < 2; i++) {
      int c = t + i * 256, r = c >> 2, cc = (c & 3) << 3;
      areg[i] = *(const v8h*)(X + (size_t)(m0 + r) * D_MODEL + k0 + cc);
    }
    #pragma unroll
    for (int i = 0; i < 2; i++) {
      int c = t + i * 256, r = c >> 3, cc = (c & 7) << 2;
      breg[i] = *(const float4*)(W + (size_t)(n0 + r) * D_MODEL + k0 + cc);
    }
  };
  load_tiles(0);

  for (int k0 = 0; k0 < D_MODEL; k0 += 32) {
    __syncthreads();
    #pragma unroll
    for (int i = 0; i < 2; i++) {
      int c = t + i * 256, r = c >> 2, cc = (c & 3) << 3;
      *(v8h*)&As[r][cc] = areg[i];
    }
    #pragma unroll
    for (int i = 0; i < 2; i++) {
      int c = t + i * 256, r = c >> 3, cc = (c & 7) << 2;
      Bs[r][cc + 0] = (half_t)breg[i].x; Bs[r][cc + 1] = (half_t)breg[i].y;
      Bs[r][cc + 2] = (half_t)breg[i].z; Bs[r][cc + 3] = (half_t)breg[i].w;
    }
    if (k0 + 32 < D_MODEL) load_tiles(k0 + 32);
    __syncthreads();

    #pragma unroll
    for (int ti = 0; ti < 2; ti++) {
      const half_t* ar = &As[wm + ti * 16 + l16][0];
      v8h alo = *(const v8h*)(ar + hh * 16);
      v8h ahi = *(const v8h*)(ar + hh * 16 + 8);
      v16h a;
      #pragma unroll
      for (int j = 0; j < 8; j++) { a[j] = alo[j]; a[8 + j] = ahi[j]; }
      #pragma unroll
      for (int tj = 0; tj < 2; tj++) {
        v16h b = *(const v16h*)(&Bs[wn + tj * 16 + l16][hh * 16]);
        acc[ti][tj] = wmma_f16(a, b, acc[ti][tj]);
      }
    }
  }

  // direct store with bias (C layout: M = r + 8*hh, N = l16)
  #pragma unroll
  for (int ti = 0; ti < 2; ti++)
    #pragma unroll
    for (int tj = 0; tj < 2; tj++)
      #pragma unroll
      for (int r = 0; r < 8; r++) {
        int grow = m0 + wm + ti * 16 + r + 8 * hh;
        int gcol = n0 + wn + tj * 16 + l16;
        Y[(size_t)grow * D_MODEL + gcol] = acc[ti][tj][r] + bias[gcol];
      }
}

// ---------------------------------------------------------------------------
extern "C" void kernel_launch(void* const* d_in, const int* in_sizes, int n_in,
                              void* d_out, int out_size, void* d_ws, size_t ws_size,
                              hipStream_t stream)
{
  const float* query = (const float*)d_in[0];
  const float* kv    = (const float*)d_in[1];
  const float* q_w   = (const float*)d_in[2];
  const float* q_b   = (const float*)d_in[3];
  const float* k_w   = (const float*)d_in[4];
  const float* k_b   = (const float*)d_in[5];
  const float* v_w   = (const float*)d_in[6];
  const float* v_b   = (const float*)d_in[7];
  const float* out_w = (const float*)d_in[8];
  const float* out_b = (const float*)d_in[9];
  float* out = (float*)d_out;

  const int B = 4, S = 1024, K = 2048;

  // workspace: qh (8MB) | kh (16MB) | vh (16MB) | ao (8MB), all f16
  half_t* qh = (half_t*)d_ws;
  half_t* kh = qh + (size_t)B * NHEAD * S * HEAD_DIM;
  half_t* vh = kh + (size_t)B * NHEAD * K * HEAD_DIM;
  half_t* ao = vh + (size_t)B * NHEAD * K * HEAD_DIM;

  dim3 blk256(256);
  proj_rope_kernel<<<dim3(B * S / 128, NHEAD), blk256, 0, stream>>>(
      query, q_w, q_b, qh, S, 2);
  proj_rope_kernel<<<dim3(B * K / 128, NHEAD), blk256, 0, stream>>>(
      kv, k_w, k_b, kh, K, 1);
  proj_rope_kernel<<<dim3(B * K / 128, NHEAD), blk256, 0, stream>>>(
      kv, v_w, v_b, vh, K, 0);
  attn_kernel<<<dim3(S / 64, NHEAD, B), dim3(128), 0, stream>>>(
      qh, kh, vh, ao, S, K);
  out_proj_kernel<<<dim3(B * S / 128, NHEAD), blk256, 0, stream>>>(
      ao, out_w, out_b, out);
}
